// WindowInterAttention_8744553414873
// MI455X (gfx1250) — compile-verified, matrix operand order checked by reading, not run on hardware
//
#include <hip/hip_runtime.h>
#include <math.h>

typedef __attribute__((ext_vector_type(16))) _Float16 v16h;
typedef __attribute__((ext_vector_type(8)))  _Float16 v8h;
typedef __attribute__((ext_vector_type(8)))  float    v8f;
typedef __attribute__((ext_vector_type(4)))  float    v4f;

// ---------- problem constants ----------
#define BB    8
#define HH    64
#define WW    64
#define CC    64
#define NW    4
#define NH    4
#define K2    9
#define DH    16
#define DD    576   // C*K*K
#define LL    4096

// ---------- workspace layout ----------
// float region (element offsets)
#define AQK_F      0          // 128*81        = 10368
#define KERN0_F    10368      // 8*4*576       = 18432
#define POOLED_F   28800      // 8*4*64        = 2048
#define ALLK_F     30848      // 8*5*576       = 23040
#define F_TOTAL    53888
// half region starts at byte offset F_TOTAL*4 = 215552 (32B aligned)
#define HREG_BYTE  215552
#define BFRAG_H    0          // 8*4*18*512    = 294912 halves
#define XPAD_H     294912     // 8*66*66*64    = 2230272 halves

// ================= kernel 0: zero the aqk accumulator =================
__global__ void k_init(float* __restrict__ aqk) {
  int i = blockIdx.x * 256 + threadIdx.x;
  if (i < 128 * 81) aqk[i] = 0.f;
}

// ================= kernel 1: pool attn 1024x1024 -> 9x9 (sums) =================
// grid: 128 matrices * 16 slabs(64 rows); block: 256 (8 waves)
// Streams 512MB once: nontemporal loads keep the L2 clean for the GEMM tiles.
__global__ __launch_bounds__(256) void k_pool_attn(const float* __restrict__ attn,
                                                   float* __restrict__ aqkSum) {
  int m    = blockIdx.x >> 4;
  int slab = blockIdx.x & 15;
  int tid  = threadIdx.x;
  int wv   = tid >> 5, lane = tid & 31;

  __shared__ float s81[81];
  if (tid < 81) s81[tid] = 0.f;
  __syncthreads();

  // row-bin setup (slab of 64 rows spans at most 2 bins)
  int r0 = slab * 64;
  int rb0 = (9 * r0) >> 10;                    // binLo(r0)
  int rEnd0   = (1024 * (rb0 + 1) + 8) / 9;    // ceil end of rb0
  bool useG1  = (rb0 < 8);
  int rb1     = useG1 ? rb0 + 1 : 8;
  int rStart1 = (1024 * rb1) / 9;

  // col-bin setup per lane (32-col chunk spans at most 2 bins)
  int c0l  = lane * 32;
  int beta = (9 * c0l) >> 10;
  int cS = (1024 * (beta + 1)) / 9;            // start of beta+1
  int cE = (1024 * (beta + 1) + 8) / 9;        // end of beta (ceil)
  int jE = cE - c0l;
  int jS = cS - c0l;

  float A00 = 0.f, A01 = 0.f, A10 = 0.f, A11 = 0.f;

  for (int i = 0; i < 8; ++i) {
    int r = r0 + wv * 8 + i;
    const v4f* rp =
        reinterpret_cast<const v4f*>(attn + ((size_t)m * 1024 + r) * 1024) + lane * 8;
    float racc0 = 0.f, racc1 = 0.f;
#pragma unroll
    for (int q = 0; q < 8; ++q) {
      v4f v = __builtin_nontemporal_load(rp + q);
      int j0 = q * 4;
      racc0 += (j0 + 0 < jE ? v[0] : 0.f);
      racc0 += (j0 + 1 < jE ? v[1] : 0.f);
      racc0 += (j0 + 2 < jE ? v[2] : 0.f);
      racc0 += (j0 + 3 < jE ? v[3] : 0.f);
      racc1 += (j0 + 0 >= jS ? v[0] : 0.f);
      racc1 += (j0 + 1 >= jS ? v[1] : 0.f);
      racc1 += (j0 + 2 >= jS ? v[2] : 0.f);
      racc1 += (j0 + 3 >= jS ? v[3] : 0.f);
    }
    bool g0 = (r < rEnd0);
    bool g1 = useG1 && (r >= rStart1);
    if (g0) { A00 += racc0; A01 += racc1; }
    if (g1) { A10 += racc0; A11 += racc1; }
  }

  int b1 = (beta + 1 < 8) ? beta + 1 : 8;
  atomicAdd(&s81[rb0 * 9 + beta], A00);
  atomicAdd(&s81[rb0 * 9 + b1],   A01);
  if (useG1) {
    atomicAdd(&s81[rb1 * 9 + beta], A10);
    atomicAdd(&s81[rb1 * 9 + b1],   A11);
  }
  __syncthreads();
  if (tid < 81) atomicAdd(&aqkSum[m * 81 + tid], s81[tid]);
}

// ================= kernel 2: pad + convert x -> f16 halo buffer =================
__global__ void k_pad_x(const float* __restrict__ x, _Float16* __restrict__ xpad) {
  size_t gid = (size_t)blockIdx.x * 256 + threadIdx.x;
  if (gid >= (size_t)BB * 66 * 66 * 64) return;
  int c = gid & 63;
  size_t t = gid >> 6;
  int wp = (int)(t % 66); t /= 66;
  int hp = (int)(t % 66);
  int b  = (int)(t / 66);
  float v = 0.f;
  if (hp >= 1 && hp <= 64 && wp >= 1 && wp <= 64)
    v = x[(((size_t)b * 64 + (hp - 1)) * 64 + (wp - 1)) * 64 + c];
  xpad[gid] = (_Float16)v;
}

// ================= kernel 3: window pooling of x (kern0 + channel means) =========
// grid: 32 (b,nw); block: 576 (c=t/9, bin=t%9)
__global__ __launch_bounds__(576) void k_pool_x(const float* __restrict__ x,
                                                float* __restrict__ kern0,
                                                float* __restrict__ pooled) {
  int bw = blockIdx.x;
  int b = bw >> 2, nw = bw & 3;
  int hb = (nw >> 1) * 32, wb = (nw & 1) * 32;
  int t = threadIdx.x;
  int c = t / 9, ij = t % 9, ky = ij / 3, kx = ij % 3;
  // torch adaptive bins for 32 -> 3: start=floor(i*32/3), end=ceil((i+1)*32/3)
  int rs = (32 * ky) / 3,     re = (32 * (ky + 1) + 2) / 3;
  int cs = (32 * kx) / 3,     ce = (32 * (kx + 1) + 2) / 3;
  float s = 0.f;
  for (int r = rs; r < re; ++r) {
    const float* row = x + ((size_t)(b * 64 + hb + r) * 64 + wb) * 64 + c;
    for (int q = cs; q < ce; ++q) s += row[(size_t)q * 64];
  }
  kern0[(size_t)bw * DD + c * 9 + ij] = s / (float)((re - rs) * (ce - cs));

  if (ij == 0) {
    float ps = 0.f;
    for (int r = 0; r < 32; ++r) {
      const float* row = x + ((size_t)(b * 64 + hb + r) * 64 + wb) * 64 + c;
      for (int q = 0; q < 32; ++q) ps += row[(size_t)q * 64];
    }
    pooled[bw * 64 + c] = ps * (1.f / 1024.f);
  }
}

// ================= kernel 4: per-batch small pipeline -> all_k =================
__global__ __launch_bounds__(256) void k_small(
    const float* __restrict__ aqkSum, const float* __restrict__ kern0,
    const float* __restrict__ pooled,
    const float* __restrict__ pvw, const float* __restrict__ pvb,
    const float* __restrict__ pow_, const float* __restrict__ pob,
    const float* __restrict__ down_w, const float* __restrict__ down_b,
    const float* __restrict__ lin1_w, const float* __restrict__ lin1_b,
    const float* __restrict__ lin2_w, const float* __restrict__ lin2_b,
    const float* __restrict__ gk_w, const float* __restrict__ gk_b,
    float* __restrict__ allk) {
  int b = blockIdx.x, t = threadIdx.x;
  __shared__ float aqk[16 * 81];
  __shared__ float k0[4 * DD];
  __shared__ float kvr[2304];
  __shared__ float k1[2304];
  __shared__ float k2s[2304];
  __shared__ float k3[2304];
  __shared__ float wwgt[4], hmid[16], wws[4];

  // finalize aqk (divide by bin areas)
  for (int idx = t; idx < 16 * 81; idx += 256) {
    int m16 = idx / 81, ij = idx % 81, i = ij / 9, j = ij % 9;
    int rs = (1024 * i) / 9,     re = (1024 * (i + 1) + 8) / 9;
    int cs = (1024 * j) / 9,     ce = (1024 * (j + 1) + 8) / 9;
    aqk[idx] = aqkSum[(b * 16 + m16) * 81 + ij] / (float)((re - rs) * (ce - cs));
  }
  for (int idx = t; idx < 4 * DD; idx += 256) k0[idx] = kern0[(size_t)b * 2304 + idx];
  __syncthreads();

  // kv = kern0 @ proj_v_w^T + b, stored as (nw,nh,k2,dh)
  for (int idx = t; idx < 2304; idx += 256) {
    int nw = idx / DD, d = idx % DD;
    float s = pvb[d];
    const float* wr = pvw + (size_t)d * DD;
    const float* kr = k0 + nw * DD;
    for (int e = 0; e < DD; ++e) s += kr[e] * wr[e];
    int nh = d / 144, rem = d % 144, dh = rem / 9, kq = rem % 9;
    kvr[((nw * 4 + nh) * 9 + kq) * 16 + dh] = s;
  }
  __syncthreads();

  // kern1 = aqk @ kv  (per (nw,nh): 9x9 @ 9x16)
  for (int idx = t; idx < 2304; idx += 256) {
    int m16 = idx / 144, rr = idx % 144, kq = rr / 16, dh = rr % 16;
    float s = 0.f;
    for (int j = 0; j < 9; ++j) s += aqk[m16 * 81 + kq * 9 + j] * kvr[(m16 * 9 + j) * 16 + dh];
    k1[(m16 * 9 + kq) * 16 + dh] = s;
  }
  __syncthreads();

  // faithful (B,NH,NW) reinterpret + transpose -> kern2 (nw, d)
  for (int idx = t; idx < 2304; idx += 256) {
    int n = idx / DD, d = idx % DD;
    int p = d / 144, rem = d % 144, dh = rem / 9, kq = rem % 9;
    k2s[idx] = k1[((p * 4 + n) * 9 + kq) * 16 + dh];
  }
  __syncthreads();

  // kern3 = kern2 @ proj_out_w^T + b
  for (int idx = t; idx < 2304; idx += 256) {
    int n = idx / DD, d = idx % DD;
    float s = pob[d];
    const float* wr = pow_ + (size_t)d * DD;
    const float* kr = k2s + n * DD;
    for (int e = 0; e < DD; ++e) s += kr[e] * wr[e];
    k3[idx] = s;
  }

  // squeeze-excite over windows
  if (t < 4) {
    float s = down_b[t];
    const float* pr = pooled + (b * 4 + t) * 64;
    for (int c = 0; c < 64; ++c) s += pr[c] * down_w[t * 64 + c];
    wwgt[t] = s;
  }
  __syncthreads();
  if (t < 16) {
    float s = lin1_b[t];
    for (int n = 0; n < 4; ++n) s += wwgt[n] * lin1_w[t * 4 + n];
    hmid[t] = 0.5f * s * (1.0f + erff(s * 0.70710678118654752f));  // exact gelu
  }
  __syncthreads();
  if (t < 4) {
    float s = lin2_b[t];
    for (int j = 0; j < 16; ++j) s += hmid[j] * lin2_w[t * 16 + j];
    wws[t] = 1.0f / (1.0f + expf(-s));
  }
  __syncthreads();

  float gb = gk_b[0];
  for (int idx = t; idx < 5 * DD; idx += 256) {
    int w5 = idx / DD, d = idx % DD;
    float v;
    if (w5 < 4) v = k3[w5 * DD + d] * wws[w5];
    else {
      v = gb;
      for (int n = 0; n < 4; ++n) v += k3[n * DD + d] * wws[n] * gk_w[n];
    }
    allk[(size_t)(b * 5 + w5) * DD + d] = v;
  }
}

// ================= kernel 5: build K2eff in WMMA B-fragment layout (f16) ========
// K2eff[b][k=tap*64+c][o] = sum_w5 fusion_w[o, w5*64+c] * all_k[b,w5,c*9+tap]
__global__ void k_bfrag(const float* __restrict__ fusion_w,
                        const float* __restrict__ allk,
                        _Float16* __restrict__ bfrag) {
  int gid = blockIdx.x * 256 + threadIdx.x;
  if (gid >= BB * DD * 64) return;
  int n = gid & 63;
  int t = gid >> 6;
  int k = t % DD;
  int b = t / DD;
  int tap = k >> 6, c = k & 63;
  float v = 0.f;
#pragma unroll
  for (int w5 = 0; w5 < 5; ++w5)
    v += fusion_w[n * 320 + w5 * 64 + c] * allk[(size_t)(b * 5 + w5) * DD + c * 9 + tap];

  int nt = n >> 4, ncol = n & 15;
  int ks = k >> 5, kr = k & 31;
  int lane, e;
  if (kr < 8)       { lane = ncol;      e = kr;      }
  else if (kr < 16) { lane = ncol + 16; e = kr - 8;  }
  else if (kr < 24) { lane = ncol;      e = kr - 8;  }
  else              { lane = ncol + 16; e = kr - 16; }
  bfrag[((size_t)(b * 4 + nt) * 18 + ks) * 512 + lane * 16 + e] = (_Float16)v;
}

// ================= kernel 6: fused conv+fusion GEMM via WMMA ====================
// Per batch: (4096 x 576) @ (576 x 64), K order = tap*64 + c.
// grid: 8 batches * 32 pixel-blocks; block: 256 (8 waves, each a 16x64 strip).
// Fully unrolled + software-pipelined: k+1's A/B fragments are issued before
// k's WMMAs; full unroll lets renaming absorb the double buffer (no v_mov,
// no WMMA->VALU hazard NOPs).
__global__ __launch_bounds__(256) void k_gemm(const _Float16* __restrict__ xpad,
                                              const _Float16* __restrict__ bfrag,
                                              const float* __restrict__ fusion_b,
                                              float* __restrict__ out) {
  int b    = blockIdx.x >> 5;
  int pblk = blockIdx.x & 31;
  int wv   = threadIdx.x >> 5;
  int lane = threadIdx.x & 31;
  int pix0 = pblk * 128 + wv * 16;

  int r  = lane & 15;
  bool hi = (lane >= 16);
  int pix = pix0 + r;
  int h = pix >> 6, w = pix & 63;

  v8f acc0 = {}, acc1 = {}, acc2 = {}, acc3 = {};
  const _Float16* bb = bfrag + (size_t)(b * 4) * 18 * 512;
  const _Float16* xb = xpad + (size_t)b * 66 * 66 * 64;
  const int coff = (hi ? 8 : 0);

  // ---- prologue: loads for ks = 0 (tap 0 => dy=0, dx=0, c0=0)
  size_t abase = (((size_t)h) * 66 + w) * 64 + coff;
  v8h alo = *(const v8h*)(xb + abase);
  v8h ahi = *(const v8h*)(xb + abase + 16);
  const _Float16* bks = bb + lane * 16;
  v16h b0 = *(const v16h*)(bks);
  v16h b1 = *(const v16h*)(bks + 18 * 512);
  v16h b2 = *(const v16h*)(bks + 2 * 18 * 512);
  v16h b3 = *(const v16h*)(bks + 3 * 18 * 512);

#pragma unroll
  for (int ks = 0; ks < 18; ++ks) {
    // ---- prefetch fragments for next k-step (redundant re-load on last iter)
    int kn = (ks < 17) ? ks + 1 : 17;
    int tapN = kn >> 1;
    int c0N  = (kn & 1) * 32;
    int dyN = tapN / 3, dxN = tapN % 3;
    size_t abaseN = (((size_t)(h + dyN)) * 66 + (w + dxN)) * 64 + c0N + coff;
    v8h aloN = *(const v8h*)(xb + abaseN);
    v8h ahiN = *(const v8h*)(xb + abaseN + 16);
    const _Float16* bksN = bb + (size_t)kn * 512 + lane * 16;
    v16h b0N = *(const v16h*)(bksN);
    v16h b1N = *(const v16h*)(bksN + 18 * 512);
    v16h b2N = *(const v16h*)(bksN + 2 * 18 * 512);
    v16h b3N = *(const v16h*)(bksN + 3 * 18 * 512);

    // ---- assemble current A fragment:
    // lanes 0-15 = K {0..7,16..23}, lanes 16-31 = K {8..15,24..31}
    v16h a;
#pragma unroll
    for (int i = 0; i < 8; ++i) { a[i] = alo[i]; a[8 + i] = ahi[i]; }

    acc0 = __builtin_amdgcn_wmma_f32_16x16x32_f16(false, a, false, b0, (short)0, acc0, false, false);
    acc1 = __builtin_amdgcn_wmma_f32_16x16x32_f16(false, a, false, b1, (short)0, acc1, false, false);
    acc2 = __builtin_amdgcn_wmma_f32_16x16x32_f16(false, a, false, b2, (short)0, acc2, false, false);
    acc3 = __builtin_amdgcn_wmma_f32_16x16x32_f16(false, a, false, b3, (short)0, acc3, false, false);

    alo = aloN; ahi = ahiN;
    b0 = b0N; b1 = b1N; b2 = b2N; b3 = b3N;
  }

  int ncol = lane & 15;
  float fb0 = fusion_b[ncol];
  float fb1 = fusion_b[16 + ncol];
  float fb2 = fusion_b[32 + ncol];
  float fb3 = fusion_b[48 + ncol];
#pragma unroll
  for (int v = 0; v < 8; ++v) {
    int row = v + (hi ? 8 : 0);
    size_t o = ((size_t)b * LL + pix0 + row) * 64;
    out[o + ncol]      = acc0[v] + fb0;
    out[o + 16 + ncol] = acc1[v] + fb1;
    out[o + 32 + ncol] = acc2[v] + fb2;
    out[o + 48 + ncol] = acc3[v] + fb3;
  }
}

// ================= host launcher =================
extern "C" void kernel_launch(void* const* d_in, const int* in_sizes, int n_in,
                              void* d_out, int out_size, void* d_ws, size_t ws_size,
                              hipStream_t stream) {
  (void)in_sizes; (void)n_in; (void)out_size; (void)ws_size;
  const float* x        = (const float*)d_in[0];
  const float* attn     = (const float*)d_in[1];
  const float* pvw      = (const float*)d_in[2];
  const float* pvb      = (const float*)d_in[3];
  const float* pow_     = (const float*)d_in[4];
  const float* pob      = (const float*)d_in[5];
  const float* down_w   = (const float*)d_in[6];
  const float* down_b   = (const float*)d_in[7];
  const float* lin1_w   = (const float*)d_in[8];
  const float* lin1_b   = (const float*)d_in[9];
  const float* lin2_w   = (const float*)d_in[10];
  const float* lin2_b   = (const float*)d_in[11];
  const float* gk_w     = (const float*)d_in[12];
  const float* gk_b     = (const float*)d_in[13];
  const float* fusion_w = (const float*)d_in[14];
  const float* fusion_b = (const float*)d_in[15];
  float* out = (float*)d_out;

  float*    wsf = (float*)d_ws;
  _Float16* wsh = (_Float16*)((char*)d_ws + HREG_BYTE);

  float* aqkSum = wsf + AQK_F;
  float* kern0  = wsf + KERN0_F;
  float* pooled = wsf + POOLED_F;
  float* allk   = wsf + ALLK_F;
  _Float16* bfrag = wsh + BFRAG_H;
  _Float16* xpad  = wsh + XPAD_H;

  k_init<<<(128 * 81 + 255) / 256, 256, 0, stream>>>(aqkSum);
  k_pool_attn<<<128 * 16, 256, 0, stream>>>(attn, aqkSum);
  k_pad_x<<<(BB * 66 * 66 * 64 + 255) / 256, 256, 0, stream>>>(x, xpad);
  k_pool_x<<<BB * NW, 576, 0, stream>>>(x, kern0, pooled);
  k_small<<<BB, 256, 0, stream>>>(aqkSum, kern0, pooled, pvw, pvb, pow_, pob,
                                  down_w, down_b, lin1_w, lin1_b, lin2_w, lin2_b,
                                  gk_w, gk_b, allk);
  k_bfrag<<<(BB * DD * 64 + 255) / 256, 256, 0, stream>>>(fusion_w, allk, bfrag);
  k_gemm<<<BB * 32, 256, 0, stream>>>(xpad, bfrag, fusion_b, out);
}
